// AttentionBlock_13288628814037
// MI455X (gfx1250) — compile-verified, matrix operand order checked by reading, not run on hardware
//
#include <hip/hip_runtime.h>
#include <hip/hip_bf16.h>

// ---------- problem constants ----------
#define BB 4
#define SS 1024
#define EE 1024
#define HH 16
#define DKK 64
#define FFD 4096
#define MM (BB * SS)   // 4096 rows

typedef __attribute__((ext_vector_type(16))) _Float16 v16h;
typedef __attribute__((ext_vector_type(8)))  _Float16 v8h;
typedef __attribute__((ext_vector_type(4)))  _Float16 v4h;
typedef __attribute__((ext_vector_type(8)))  float    v8f;

#define DEVINL __device__ __forceinline__

DEVINL v8f wmma_f16(v16h a, v16h b, v8f c) {
  return __builtin_amdgcn_wmma_f32_16x16x32_f16(
      /*neg_a=*/false, a, /*neg_b=*/false, b,
      /*c_mod=*/(short)0, c, /*reuse_a=*/false, /*reuse_b=*/false);
}

DEVINL v8f zero8() {
  v8f z;
#pragma unroll
  for (int i = 0; i < 8; ++i) z[i] = 0.0f;
  return z;
}

// Build a v16h A/B operand from two contiguous 16-byte chunks
// (chunk0 = contraction indices base..base+7, chunk1 = base+16..base+23).
DEVINL v16h ld16(const _Float16* p0, const _Float16* p1) {
  union { v16h v; v8h h[2]; } u;
  u.h[0] = *(const v8h*)p0;
  u.h[1] = *(const v8h*)p1;
  return u.v;
}

DEVINL float sigm(float v) { return 1.0f / (1.0f + __expf(-v)); }

// LDS byte offset of a __shared__ object = low 32 bits of its generic pointer
// (LDS aperture: addr[31:0] is the workgroup-relative LDS address, ISA 10.2).
DEVINL unsigned lds_off(const void* p) { return (unsigned)(size_t)p; }

// CDNA5 async memory->LDS copy, tracked with ASYNCcnt (ISA 10. / 15.18.3).
DEVINL void async_ld_b128(unsigned dst_lds, const void* gaddr) {
  asm volatile("global_load_async_to_lds_b128 %0, %1, off"
               :: "v"(dst_lds), "v"(gaddr)
               : "memory");
}
DEVINL void async_wait0() {
  asm volatile("s_wait_asynccnt 0x0" ::: "memory");
}

// ---------------------------------------------------------------------------
// f32 -> f16 conversion (vectorized by 4)
// ---------------------------------------------------------------------------
__global__ __launch_bounds__(256) void cvt_f16_kernel(
    const float* __restrict__ in, _Float16* __restrict__ out, int n4) {
  int i = blockIdx.x * 256 + threadIdx.x;
  if (i < n4) {
    float4 f = ((const float4*)in)[i];
    v4h o;
    o[0] = (_Float16)f.x; o[1] = (_Float16)f.y;
    o[2] = (_Float16)f.z; o[3] = (_Float16)f.w;
    ((v4h*)out)[i] = o;
  }
}

// ---------------------------------------------------------------------------
// pos_bias[q,k] = mean_c pos_encoding[q,k,c], c in 0..7
// ---------------------------------------------------------------------------
__global__ __launch_bounds__(256) void posbias_kernel(
    const float* __restrict__ pe, float* __restrict__ pb) {
  int i = blockIdx.x * 256 + threadIdx.x;  // i < S*S
  const float4* p = (const float4*)(pe + (size_t)i * 8);
  float4 a = p[0], b = p[1];
  pb[i] = (a.x + a.y + a.z + a.w + b.x + b.y + b.z + b.w) * 0.125f;
}

// ---------------------------------------------------------------------------
// WMMA GEMM: C[M,N] = A[M,K](f16) @ W[K,N](f16) + bias
// Block tile 128x128, 8 waves (2x4), wave tile 64x32, K-step 32.
// A tiles double-buffered in LDS, filled by global_load_async_to_lds_b128
// issued one iteration ahead (latency hidden behind WMMA work).
// W transposed to [n][k] during staging so both operands are two contiguous
// 16B LDS reads per the documented WMMA A/B VGPR layouts.
// Output: f16 if Ch != nullptr else f32 to Cf.
// ---------------------------------------------------------------------------
__global__ __launch_bounds__(256) void wmma_gemm_kernel(
    const _Float16* __restrict__ A, const _Float16* __restrict__ W,
    const float* __restrict__ bias, float* __restrict__ Cf,
    _Float16* __restrict__ Ch, int M, int N, int K) {
  __shared__ _Float16 lsA[2][128 * 32];  // [buf][row][k]
  __shared__ _Float16 lsB[128 * 32];     // [n][k]

  const int tid  = threadIdx.x;
  const int lane = tid & 31, wid = tid >> 5;
  const int l16  = lane & 15, hi = lane >> 4;
  const int b8   = hi * 8;
  const int m0 = blockIdx.y * 128, n0 = blockIdx.x * 128;
  const int wr = wid >> 2, wc = wid & 3;   // 2 x 4 wave grid

  v8f acc[4][2];
#pragma unroll
  for (int m = 0; m < 4; ++m)
#pragma unroll
    for (int n = 0; n < 2; ++n) acc[m][n] = zero8();

  // staging assignments
  const int arow = tid >> 1, aseg = tid & 1;     // A: 128 rows x 2 segs of 16h
  const int bk = tid >> 3, bseg = tid & 7;       // W: 32 k-rows x 8 segs of 16h
  const _Float16* gA = A + (size_t)(m0 + arow) * K + aseg * 16;
  const _Float16* gW = W + (size_t)bk * N + n0 + bseg * 16;
  const unsigned aoff = lds_off(&lsA[0][0]) + arow * 64 + aseg * 32;  // bytes

  // preload A tile 0 (async, no VGPR round trip)
  async_ld_b128(aoff, gA);
  async_ld_b128(aoff + 16, gA + 8);

  for (int k0 = 0; k0 < K; k0 += 32) {
    const int cur = (k0 >> 5) & 1;
    v16h wv = *(const v16h*)(gW + (size_t)k0 * N);

    async_wait0();       // own async stores into lsA[cur] have landed
    __syncthreads();     // everyone's landed; previous-tile readers done

    {
      alignas(32) _Float16 tmp[16];
      *(v16h*)tmp = wv;
#pragma unroll
      for (int i = 0; i < 16; ++i)          // transpose into [n][k]
        lsB[(bseg * 16 + i) * 32 + bk] = tmp[i];
    }
    if (k0 + 32 < K) {   // async-prefetch next A tile into the other buffer
      const unsigned noff = aoff + (cur ^ 1) * (128 * 32 * 2);
      async_ld_b128(noff, gA + k0 + 32);
      async_ld_b128(noff + 16, gA + k0 + 40);
      __builtin_prefetch(gW + (size_t)(k0 + 32) * N, 0, 3);
    }
    __syncthreads();

    v16h bv[2];
#pragma unroll
    for (int n = 0; n < 2; ++n) {
      const _Float16* bp = &lsB[(wc * 32 + n * 16 + l16) * 32];
      bv[n] = ld16(bp + b8, bp + b8 + 16);
    }
#pragma unroll
    for (int m = 0; m < 4; ++m) {
      const _Float16* ap = &lsA[cur][(wr * 64 + m * 16 + l16) * 32];
      v16h am = ld16(ap + b8, ap + b8 + 16);
      acc[m][0] = wmma_f16(am, bv[0], acc[m][0]);
      acc[m][1] = wmma_f16(am, bv[1], acc[m][1]);
    }
  }

  float bcol[2];
#pragma unroll
  for (int n = 0; n < 2; ++n) bcol[n] = bias[n0 + wc * 32 + n * 16 + l16];

#pragma unroll
  for (int m = 0; m < 4; ++m)
#pragma unroll
    for (int n = 0; n < 2; ++n) {
      const int c = n0 + wc * 32 + n * 16 + l16;
      const int rbase = m0 + wr * 64 + m * 16 + b8;
#pragma unroll
      for (int j = 0; j < 8; ++j) {
        float v = acc[m][n][j] + bcol[n];
        size_t off = (size_t)(rbase + j) * N + c;
        if (Ch) Ch[off] = (_Float16)v;
        else    Cf[off] = v;
      }
    }
}

// ---------------------------------------------------------------------------
// Fused gated-FFN GEMM: Out = f16( (A@W1+b1) * sigmoid(A@W2+b2) )
// Block tile 128x64, 8 waves (4x2), wave tile 32x32, two accumulator sets.
// A tiles double-buffered via async-to-LDS, one iteration ahead.
// ---------------------------------------------------------------------------
__global__ __launch_bounds__(256) void wmma_gemm_gated_kernel(
    const _Float16* __restrict__ A, const _Float16* __restrict__ W1,
    const _Float16* __restrict__ W2, const float* __restrict__ b1,
    const float* __restrict__ b2, _Float16* __restrict__ Out,
    int M, int N, int K) {
  __shared__ _Float16 lsA[2][128 * 32];
  __shared__ _Float16 lsB1[64 * 32];
  __shared__ _Float16 lsB2[64 * 32];

  const int tid  = threadIdx.x;
  const int lane = tid & 31, wid = tid >> 5;
  const int l16  = lane & 15, hi = lane >> 4;
  const int b8   = hi * 8;
  const int m0 = blockIdx.y * 128, n0 = blockIdx.x * 64;
  const int wr = wid >> 1, wc = wid & 1;   // 4 x 2 wave grid

  v8f a1[2][2], a2[2][2];
#pragma unroll
  for (int m = 0; m < 2; ++m)
#pragma unroll
    for (int n = 0; n < 2; ++n) { a1[m][n] = zero8(); a2[m][n] = zero8(); }

  const int arow = tid >> 1, aseg = tid & 1;
  const int t2 = tid & 127;
  const int bk = t2 >> 2, bseg = t2 & 3;         // 32 k-rows x 4 segs of 16h
  const _Float16* gA = A + (size_t)(m0 + arow) * K + aseg * 16;
  const _Float16* gW =
      (tid < 128 ? W1 : W2) + (size_t)bk * N + n0 + bseg * 16;
  _Float16* lsB = (tid < 128) ? lsB1 : lsB2;
  const unsigned aoff = lds_off(&lsA[0][0]) + arow * 64 + aseg * 32;

  async_ld_b128(aoff, gA);
  async_ld_b128(aoff + 16, gA + 8);

  for (int k0 = 0; k0 < K; k0 += 32) {
    const int cur = (k0 >> 5) & 1;
    v16h wv = *(const v16h*)(gW + (size_t)k0 * N);

    async_wait0();
    __syncthreads();

    {
      alignas(32) _Float16 tmp[16];
      *(v16h*)tmp = wv;
#pragma unroll
      for (int i = 0; i < 16; ++i)
        lsB[(bseg * 16 + i) * 32 + bk] = tmp[i];
    }
    if (k0 + 32 < K) {
      const unsigned noff = aoff + (cur ^ 1) * (128 * 32 * 2);
      async_ld_b128(noff, gA + k0 + 32);
      async_ld_b128(noff + 16, gA + k0 + 40);
      __builtin_prefetch(gW + (size_t)(k0 + 32) * N, 0, 3);
    }
    __syncthreads();

    v16h bv1[2], bv2[2];
#pragma unroll
    for (int n = 0; n < 2; ++n) {
      const int col = (wc * 32 + n * 16 + l16) * 32;
      bv1[n] = ld16(&lsB1[col + b8], &lsB1[col + b8 + 16]);
      bv2[n] = ld16(&lsB2[col + b8], &lsB2[col + b8 + 16]);
    }
#pragma unroll
    for (int m = 0; m < 2; ++m) {
      const _Float16* ap = &lsA[cur][(wr * 32 + m * 16 + l16) * 32];
      v16h am = ld16(ap + b8, ap + b8 + 16);
#pragma unroll
      for (int n = 0; n < 2; ++n) {
        a1[m][n] = wmma_f16(am, bv1[n], a1[m][n]);
        a2[m][n] = wmma_f16(am, bv2[n], a2[m][n]);
      }
    }
  }

#pragma unroll
  for (int m = 0; m < 2; ++m)
#pragma unroll
    for (int n = 0; n < 2; ++n) {
      const int c = n0 + wc * 32 + n * 16 + l16;
      const float bb1 = b1[c], bb2 = b2[c];
      const int rbase = m0 + wr * 32 + m * 16 + b8;
#pragma unroll
      for (int j = 0; j < 8; ++j) {
        float v = (a1[m][n][j] + bb1) * sigm(a2[m][n][j] + bb2);
        Out[(size_t)(rbase + j) * N + c] = (_Float16)v;
      }
    }
}

// ---------------------------------------------------------------------------
// Flash attention. Grid (S/128, H, B); 8 waves, each owns 16 query rows.
// scores = (Q·K^T)/8 + pos_bias, online softmax, ctx = P·V. All mm via WMMA.
// K tile staged via async-to-LDS; V goes through VGPRs (needs transpose).
// ---------------------------------------------------------------------------
__global__ __launch_bounds__(256) void flash_attn_kernel(
    const _Float16* __restrict__ Q, const _Float16* __restrict__ Kh,
    const _Float16* __restrict__ Vh, const float* __restrict__ PB,
    _Float16* __restrict__ O) {
  __shared__ _Float16 lsK[32 * 64];        // [key][d]
  __shared__ _Float16 lsVt[64 * 32];       // [d][key]
  __shared__ _Float16 lsP[8][16 * 32];     // per-wave probs [qrow][key]

  const int tid  = threadIdx.x;
  const int lane = tid & 31, wid = tid >> 5;
  const int l16  = lane & 15, hi = lane >> 4;
  const int b8   = hi * 8;
  const int h = blockIdx.y, bb = blockIdx.z;
  const int q0 = blockIdx.x * 128 + wid * 16;
  const size_t rs = (size_t)HH * DKK;      // 1024 token row stride

  // Q A-operands (16 rows x 64 d), two K=32 chunks
  v16h qa[2];
  {
    const _Float16* qp = Q + ((size_t)bb * SS + q0 + l16) * rs + h * DKK;
    qa[0] = ld16(qp + b8,      qp + b8 + 16);
    qa[1] = ld16(qp + 32 + b8, qp + 32 + b8 + 16);
  }

  v8f o[4];
#pragma unroll
  for (int n = 0; n < 4; ++n) o[n] = zero8();
  float mrow[8], srow[8];
#pragma unroll
  for (int j = 0; j < 8; ++j) { mrow[j] = -1e30f; srow[j] = 0.0f; }

  const int skey = tid >> 3, sseg = tid & 7;   // stage 32 keys x 8 d-segments
  const unsigned koff = lds_off(&lsK[0]) + skey * 128 + sseg * 16;

  for (int kb = 0; kb < SS; kb += 32) {
    const size_t tok = ((size_t)bb * SS + kb + skey) * rs + h * DKK + sseg * 8;
    v8h vv = *(const v8h*)(Vh + tok);

    __syncthreads();                       // previous-tile readers done
    async_ld_b128(koff, Kh + tok);         // K tile: memory -> LDS direct
    {
      alignas(16) _Float16 vt[8];
      *(v8h*)vt = vv;
#pragma unroll
      for (int i = 0; i < 8; ++i)
        lsVt[(sseg * 8 + i) * 32 + skey] = vt[i];   // transpose to [d][key]
    }
    async_wait0();
    __syncthreads();

    // scores: two 16x16 key tiles, contraction over d=64 (2 WMMAs each)
    v8f sc[2];
    sc[0] = zero8(); sc[1] = zero8();
#pragma unroll
    for (int ks = 0; ks < 2; ++ks) {
      const _Float16* kr = &lsK[(ks * 16 + l16) * 64];
      sc[ks] = wmma_f16(qa[0], ld16(kr + b8,      kr + b8 + 16),      sc[ks]);
      sc[ks] = wmma_f16(qa[1], ld16(kr + 32 + b8, kr + 32 + b8 + 16), sc[ks]);
    }

    // online softmax over the 32-key chunk
#pragma unroll
    for (int j = 0; j < 8; ++j) {
      const int qr = q0 + b8 + j;
      float v0 = sc[0][j] * 0.125f + PB[(size_t)qr * SS + kb + l16];
      float v1 = sc[1][j] * 0.125f + PB[(size_t)qr * SS + kb + 16 + l16];
      float ml = fmaxf(v0, v1);
#pragma unroll
      for (int off = 1; off < 16; off <<= 1)
        ml = fmaxf(ml, __shfl_xor(ml, off, 32));
      float mn = fmaxf(mrow[j], ml);
      float alpha = __expf(mrow[j] - mn);
      float p0 = __expf(v0 - mn), p1 = __expf(v1 - mn);
      float ps = p0 + p1;
#pragma unroll
      for (int off = 1; off < 16; off <<= 1) ps += __shfl_xor(ps, off, 32);
      srow[j] = srow[j] * alpha + ps;
      mrow[j] = mn;
#pragma unroll
      for (int n = 0; n < 4; ++n) o[n][j] *= alpha;
      lsP[wid][(b8 + j) * 32 + l16]      = (_Float16)p0;
      lsP[wid][(b8 + j) * 32 + 16 + l16] = (_Float16)p1;
    }

    // ctx += P(16x32) @ V(32x64): 4 WMMAs over d-subtiles
    const _Float16* pr = &lsP[wid][l16 * 32];
    v16h pa = ld16(pr + b8, pr + b8 + 16);
#pragma unroll
    for (int ds = 0; ds < 4; ++ds) {
      const _Float16* vr = &lsVt[(ds * 16 + l16) * 32];
      o[ds] = wmma_f16(pa, ld16(vr + b8, vr + b8 + 16), o[ds]);
    }
  }

#pragma unroll
  for (int j = 0; j < 8; ++j) {
    const float inv = 1.0f / srow[j];
    const int qr = q0 + b8 + j;
    _Float16* op = O + ((size_t)bb * SS + qr) * rs + h * DKK;
#pragma unroll
    for (int ds = 0; ds < 4; ++ds)
      op[ds * 16 + l16] = (_Float16)(o[ds][j] * inv);
  }
}

// ---------------------------------------------------------------------------
// Fused residual (+gate) + LayerNorm. One row (E=1024) per block, 256 thr.
// h = x + sigmoid(glog)*addv   (or x + addv if glog==nullptr)
// y = LN(h)*gamma + beta  -> y32 (and f16 copy y16 if non-null)
// ---------------------------------------------------------------------------
__global__ __launch_bounds__(256) void fused_ln_kernel(
    const float* __restrict__ xa, const float* __restrict__ glog,
    const float* __restrict__ addv, const float* __restrict__ gamma,
    const float* __restrict__ beta, float* __restrict__ y32,
    _Float16* __restrict__ y16) {
  __shared__ float s1[256], s2[256];
  const int tid = threadIdx.x;
  const size_t base = (size_t)blockIdx.x * EE;

  float hv[4], ssum = 0.0f, ssq = 0.0f;
#pragma unroll
  for (int i = 0; i < 4; ++i) {
    const int idx = i * 256 + tid;
    float a = addv[base + idx];
    if (glog) a *= sigm(glog[base + idx]);
    float v = xa[base + idx] + a;
    hv[i] = v;
    ssum += v;
    ssq  += v * v;
  }
  s1[tid] = ssum; s2[tid] = ssq;
  __syncthreads();
  for (int s = 128; s > 0; s >>= 1) {
    if (tid < s) { s1[tid] += s1[tid + s]; s2[tid] += s2[tid + s]; }
    __syncthreads();
  }
  const float mean = s1[0] * (1.0f / EE);
  const float var  = s2[0] * (1.0f / EE) - mean * mean;
  const float rstd = rsqrtf(var + 1e-6f);

#pragma unroll
  for (int i = 0; i < 4; ++i) {
    const int idx = i * 256 + tid;
    float y = (hv[i] - mean) * rstd * gamma[idx] + beta[idx];
    y32[base + idx] = y;
    if (y16) y16[base + idx] = (_Float16)y;
  }
}

// ---------------------------------------------------------------------------
// Host orchestration
// ---------------------------------------------------------------------------
extern "C" void kernel_launch(void* const* d_in, const int* in_sizes, int n_in,
                              void* d_out, int out_size, void* d_ws,
                              size_t ws_size, hipStream_t stream) {
  const float* x    = (const float*)d_in[0];
  const float* pe   = (const float*)d_in[1];
  const float* Wq   = (const float*)d_in[2];
  const float* bq   = (const float*)d_in[3];
  const float* Wk   = (const float*)d_in[4];
  const float* bk   = (const float*)d_in[5];
  const float* Wv   = (const float*)d_in[6];
  const float* bv   = (const float*)d_in[7];
  const float* Wo   = (const float*)d_in[8];
  const float* bo   = (const float*)d_in[9];
  const float* Wg   = (const float*)d_in[10];
  const float* bg   = (const float*)d_in[11];
  const float* Wf1  = (const float*)d_in[12];
  const float* bf1  = (const float*)d_in[13];
  const float* Wfg  = (const float*)d_in[14];
  const float* bfg  = (const float*)d_in[15];
  const float* Wf2  = (const float*)d_in[16];
  const float* bf2  = (const float*)d_in[17];
  const float* ln1g = (const float*)d_in[18];
  const float* ln1b = (const float*)d_in[19];
  const float* ln2g = (const float*)d_in[20];
  const float* ln2b = (const float*)d_in[21];

  char* ws = (char*)d_ws;
  auto at = [&](size_t mb) { return (void*)(ws + mb * (size_t)(1 << 20)); };

  // bump allocation (MB offsets); high-water ~134 MB
  _Float16* xh   = (_Float16*)at(0);    // 8 MB
  _Float16* wqh  = (_Float16*)at(8);    // 2 MB
  _Float16* wkh  = (_Float16*)at(10);
  _Float16* wvh  = (_Float16*)at(12);
  _Float16* woh  = (_Float16*)at(14);
  _Float16* wgh  = (_Float16*)at(16);
  _Float16* wf1h = (_Float16*)at(18);   // 8 MB
  _Float16* wfgh = (_Float16*)at(26);   // 8 MB
  _Float16* wf2h = (_Float16*)at(34);   // 8 MB
  float*    pb   = (float*)at(42);      // 4 MB
  _Float16* qh   = (_Float16*)at(46);   // 8 MB
  _Float16* kh   = (_Float16*)at(54);   // 8 MB
  _Float16* vh   = (_Float16*)at(62);   // 8 MB
  _Float16* ctxh = (_Float16*)at(70);   // 8 MB
  float*    glog = (float*)at(78);      // 16 MB
  float*    attn = (float*)at(94);      // 16 MB
  float*    x1   = (float*)at(110);     // 16 MB
  _Float16* x1h  = (_Float16*)at(126);  // 8 MB
  _Float16* ffh  = (_Float16*)at(46);   // 32 MB, reuses dead q/k/v/ctx
  float*    ff2  = (float*)at(78);      // 16 MB, reuses dead glog

  auto cvt = [&](const float* src, _Float16* dst, int n) {
    int n4 = n / 4;
    cvt_f16_kernel<<<(n4 + 255) / 256, 256, 0, stream>>>(src, dst, n4);
  };
  cvt(x,   xh,   MM * EE);
  cvt(Wq,  wqh,  EE * EE);
  cvt(Wk,  wkh,  EE * EE);
  cvt(Wv,  wvh,  EE * EE);
  cvt(Wo,  woh,  EE * EE);
  cvt(Wg,  wgh,  EE * EE);
  cvt(Wf1, wf1h, EE * FFD);
  cvt(Wfg, wfgh, EE * FFD);
  cvt(Wf2, wf2h, FFD * EE);

  posbias_kernel<<<(SS * SS) / 256, 256, 0, stream>>>(pe, pb);

  dim3 gE(EE / 128, MM / 128);  // (8, 32)
  wmma_gemm_kernel<<<gE, 256, 0, stream>>>(xh, wqh, bq, nullptr, qh, MM, EE, EE);
  wmma_gemm_kernel<<<gE, 256, 0, stream>>>(xh, wkh, bk, nullptr, kh, MM, EE, EE);
  wmma_gemm_kernel<<<gE, 256, 0, stream>>>(xh, wvh, bv, nullptr, vh, MM, EE, EE);
  wmma_gemm_kernel<<<gE, 256, 0, stream>>>(xh, wgh, bg, glog, nullptr, MM, EE, EE);

  flash_attn_kernel<<<dim3(SS / 128, HH, BB), 256, 0, stream>>>(qh, kh, vh, pb, ctxh);

  wmma_gemm_kernel<<<gE, 256, 0, stream>>>(ctxh, woh, bo, attn, nullptr, MM, EE, EE);

  fused_ln_kernel<<<MM, 256, 0, stream>>>(x, glog, attn, ln1g, ln1b, x1, x1h);

  wmma_gemm_gated_kernel<<<dim3(FFD / 64, MM / 128), 256, 0, stream>>>(
      x1h, wf1h, wfgh, bf1, bfg, ffh, MM, FFD, EE);

  wmma_gemm_kernel<<<gE, 256, 0, stream>>>(ffh, wf2h, bf2, ff2, nullptr, MM, EE, FFD);

  fused_ln_kernel<<<MM, 256, 0, stream>>>(x1, nullptr, ff2, ln2g, ln2b,
                                          (float*)d_out, nullptr);
}
